// MultiQueryAttention_48902497632779
// MI455X (gfx1250) — compile-verified, matrix operand order checked by reading, not run on hardware
//
#include <hip/hip_runtime.h>
#include <hip/hip_bf16.h>

// ---- problem constants (match reference) ----
#define D_MODEL  2048
#define N_HEADS  16
#define HEAD_DIM 128
#define BSZ      2
#define TSEQ     2048
#define ROWS     (BSZ * TSEQ)   // 4096 token rows

typedef __attribute__((ext_vector_type(16))) _Float16 v16h;
typedef __attribute__((ext_vector_type(8)))  _Float16 h8;
typedef __attribute__((ext_vector_type(8)))  float    v8f;

// D = A(16x32 f16) * B(32x16 f16) + C(16x16 f32)
__device__ __forceinline__ v8f wmma16x16x32(v16h a, v16h b, v8f c) {
  return __builtin_amdgcn_wmma_f32_16x16x32_f16(
      /*neg_a=*/false, a, /*neg_b=*/false, b,
      /*c_mod=*/(short)0, c, /*reuse_a=*/false, /*reuse_b=*/false);
}

// Load one 16x32 f16 fragment given the lane's already-offset row pointer
// (p points at row*ld + k0, where k0 includes the +8 shift for lanes 16..31).
__device__ __forceinline__ v16h load_frag_p(const _Float16* __restrict__ p) {
  h8 lo = *(const h8*)(p);
  h8 hi = *(const h8*)(p + 16);
  v16h f;
#pragma unroll
  for (int i = 0; i < 8; ++i) { f[i] = lo[i]; f[i + 8] = hi[i]; }
  return f;
}

__device__ __forceinline__ v16h load_frag(const _Float16* __restrict__ base,
                                          int ld, int row, int k0) {
  return load_frag_p(base + (size_t)row * (size_t)ld + (size_t)k0);
}

// ------------------- conversion kernels -------------------
__global__ void convert_f32_to_f16(const float* __restrict__ in,
                                   _Float16* __restrict__ out, int n) {
  int i = blockIdx.x * blockDim.x + threadIdx.x;
  if (i < n) out[i] = (_Float16)in[i];
}

// W [K][N] f32 row-major  ->  WT [N][K] f16 row-major (coalesced read)
__global__ void convert_transpose_f16(const float* __restrict__ W,
                                      _Float16* __restrict__ WT,
                                      int K, int N) {
  int i = blockIdx.x * blockDim.x + threadIdx.x;
  if (i >= K * N) return;
  int k = i / N, n = i % N;
  WT[(size_t)n * K + k] = (_Float16)W[i];
}

// ------------------- register-blocked WMMA GEMM -------------------
// C[M][N] = A[M][K] (f16 row-major) * BT[N][K]^T (f16, pre-transposed).
// One wave computes a 64x32 C block as a 4x2 grid of 16x16 WMMA tiles:
// per 32-wide k-step: 8 A-frag + 4 B-frag b128 loads -> 8 WMMAs.
// Register budget: 64 acc + 32 A-frag + 16 B-frag + addrs (~140 VGPRs),
// fits without spilling and keeps 2+ waves/SIMD resident.
// mode 0: store f16 as q layout [B][H][T][hd]
// mode 1: store f16 row-major [M][N]                (k projection)
// mode 2: store f16 transposed-per-batch [B][hd][T] (v projection)
// mode 3: store f32 row-major [M][N]                (final output)
__global__ void __launch_bounds__(128, 1)
gemm_wmma_blocked(const _Float16* __restrict__ A,
                  const _Float16* __restrict__ BT,
                  void* __restrict__ Cout,
                  int M, int N, int K, int mode) {
  int wave = blockIdx.x * (blockDim.x >> 5) + (threadIdx.x >> 5);
  int lane = threadIdx.x & 31;
  int tilesN = N >> 5;                 // 32-wide N blocks
  int tm = wave / tilesN;
  int tn = wave % tilesN;
  if (tm * 64 >= M) return;            // wave-uniform: EXEC stays all-1s

  int lrow   = lane & 15;
  int kshift = (lane & 16) ? 8 : 0;

  // per-fragment row pointers, stepped by k each iteration
  const _Float16* ap[4];
  const _Float16* bp[2];
#pragma unroll
  for (int i = 0; i < 4; ++i)
    ap[i] = A + (size_t)(tm * 64 + i * 16 + lrow) * K + kshift;
#pragma unroll
  for (int j = 0; j < 2; ++j)
    bp[j] = BT + (size_t)(tn * 32 + j * 16 + lrow) * K + kshift;

  v8f acc[4][2];
#pragma unroll
  for (int i = 0; i < 4; ++i)
#pragma unroll
    for (int j = 0; j < 2; ++j) acc[i][j] = (v8f){};

  for (int k = 0; k < K; k += 32) {
    v16h af[4], bf[2];
#pragma unroll
    for (int i = 0; i < 4; ++i) {
      __builtin_prefetch(ap[i] + k + 64, 0, 1);
      af[i] = load_frag_p(ap[i] + k);
    }
#pragma unroll
    for (int j = 0; j < 2; ++j)
      bf[j] = load_frag_p(bp[j] + k);
#pragma unroll
    for (int i = 0; i < 4; ++i)
#pragma unroll
      for (int j = 0; j < 2; ++j)
        acc[i][j] = wmma16x16x32(af[i], bf[j], acc[i][j]);
  }

  // C fragment layout: VGPR r, lanes 0-15 -> M=r, lanes 16-31 -> M=r+8
  int m0 = (lane & 16) ? 8 : 0;
#pragma unroll
  for (int i = 0; i < 4; ++i) {
#pragma unroll
    for (int j = 0; j < 2; ++j) {
      int col = tn * 32 + j * 16 + lrow;
#pragma unroll
      for (int r = 0; r < 8; ++r) {
        int   row = tm * 64 + i * 16 + m0 + r;
        float v   = acc[i][j][r];
        if (mode == 0) {            // q: [B][H][T][hd]
          int b = row / TSEQ, t = row % TSEQ;
          int h = col / HEAD_DIM, d = col % HEAD_DIM;
          ((_Float16*)Cout)[(((size_t)b * N_HEADS + h) * TSEQ + t) * HEAD_DIM + d] =
              (_Float16)v;
        } else if (mode == 1) {     // k: [B*T][hd]
          ((_Float16*)Cout)[(size_t)row * N + col] = (_Float16)v;
        } else if (mode == 2) {     // v^T: [B][hd][T]
          int b = row / TSEQ, t = row % TSEQ;
          ((_Float16*)Cout)[((size_t)b * HEAD_DIM + col) * TSEQ + t] = (_Float16)v;
        } else {                    // final: f32 [B*T][D]
          ((float*)Cout)[(size_t)row * N + col] = v;
        }
      }
    }
  }
}

// ------------------- flash MQA attention -------------------
// One wave per (batch, head, 16-query tile). Shared single KV head.
__global__ void mqa_attn_kernel(const _Float16* __restrict__ qh,
                                const _Float16* __restrict__ kh,
                                const _Float16* __restrict__ vT,
                                _Float16* __restrict__ attnh) {
  __shared__ _Float16 plds[4][16][32];          // per-wave 16x32 prob tile

  int wave = blockIdx.x * (blockDim.x >> 5) + (threadIdx.x >> 5);
  int lane = threadIdx.x & 31;
  int w    = (threadIdx.x >> 5) & 3;

  const int QT = TSEQ / 16;                     // 128 query tiles
  int b   = wave / (N_HEADS * QT);
  int rem = wave % (N_HEADS * QT);
  int h   = rem / QT;
  int qt  = rem % QT;
  if (b >= BSZ) return;                         // wave-uniform

  int qbase  = qt * 16;
  int lrow   = lane & 15;
  int kshift = (lane & 16) ? 8 : 0;
  int m0     = (lane & 16) ? 8 : 0;

  const _Float16* qptr = qh + ((size_t)b * N_HEADS + h) * TSEQ * HEAD_DIM;
  const _Float16* kptr = kh + (size_t)b * TSEQ * HEAD_DIM;
  const _Float16* vptr = vT + (size_t)b * HEAD_DIM * TSEQ;

  // preload q fragments over hd=128 (4 K-steps of 32)
  v16h qf[4];
#pragma unroll
  for (int kk = 0; kk < 4; ++kk)
    qf[kk] = load_frag(qptr, HEAD_DIM, qbase + lrow, kk * 32 + kshift);

  v8f   o[8];
  float mrow[8], lsum[8];
#pragma unroll
  for (int r = 0; r < 8; ++r) { o[r] = (v8f){}; mrow[r] = -1e30f; lsum[r] = 0.f; }

  const float scale = 0.08838834764831843f;     // 1/sqrt(128)
  int nkb = (qbase + 15) / 32 + 1;              // causal: key blocks of 32

  for (int kb = 0; kb < nkb; ++kb) {
    int kb0 = kb * 32;
    // ---- scores: two 16x16 tiles over 32 keys ----
    v8f s0 = {}, s1 = {};
#pragma unroll
    for (int kk = 0; kk < 4; ++kk) {
      v16h b0 = load_frag(kptr, HEAD_DIM, kb0 + lrow,      kk * 32 + kshift);
      v16h b1 = load_frag(kptr, HEAD_DIM, kb0 + 16 + lrow, kk * 32 + kshift);
      s0 = wmma16x16x32(qf[kk], b0, s0);
      s1 = wmma16x16x32(qf[kk], b1, s1);
    }
    // ---- causal mask + scale + online softmax ----
    int key0 = kb0 + lrow, key1 = kb0 + 16 + lrow;
#pragma unroll
    for (int r = 0; r < 8; ++r) {
      int q = qbase + m0 + r;
      float a = (key0 <= q) ? s0[r] * scale : -1e30f;
      float c = (key1 <= q) ? s1[r] * scale : -1e30f;
      float t = fmaxf(a, c);
      t = fmaxf(t, __shfl_xor(t, 1, 32));
      t = fmaxf(t, __shfl_xor(t, 2, 32));
      t = fmaxf(t, __shfl_xor(t, 4, 32));
      t = fmaxf(t, __shfl_xor(t, 8, 32));       // row max over 16-lane half
      float mnew = fmaxf(mrow[r], t);
      float corr = __expf(mrow[r] - mnew);
      mrow[r] = mnew;
      a = __expf(a - mnew);
      c = __expf(c - mnew);
      float rs = a + c;
      rs += __shfl_xor(rs, 1, 32);
      rs += __shfl_xor(rs, 2, 32);
      rs += __shfl_xor(rs, 4, 32);
      rs += __shfl_xor(rs, 8, 32);
      lsum[r] = lsum[r] * corr + rs;
#pragma unroll
      for (int j = 0; j < 8; ++j) o[j][r] *= corr;   // rescale row r of O
      s0[r] = a; s1[r] = c;
    }
    // ---- reshape probs: C-layout -> A-fragment via per-wave LDS tile ----
#pragma unroll
    for (int r = 0; r < 8; ++r) {
      plds[w][m0 + r][lrow]      = (_Float16)s0[r];
      plds[w][m0 + r][16 + lrow] = (_Float16)s1[r];
    }
    asm volatile("s_wait_dscnt 0" ::: "memory");     // wave-local DS RAW
    v16h pa;
    {
      const _Float16* pl = &plds[w][lrow][kshift];
      h8 lo = *(const h8*)(pl);
      h8 hi = *(const h8*)(pl + 16);
#pragma unroll
      for (int i = 0; i < 8; ++i) { pa[i] = lo[i]; pa[i + 8] = hi[i]; }
    }
    // ---- O += P (16x32) * V (32x128), via pre-transposed vT [hd][T] ----
#pragma unroll
    for (int j = 0; j < 8; ++j) {
      v16h vb = load_frag(vptr, TSEQ, j * 16 + lrow, kb0 + kshift);
      o[j] = wmma16x16x32(pa, vb, o[j]);
    }
  }

  // ---- normalize and store as f16 [B*T][D] with head offset ----
#pragma unroll
  for (int r = 0; r < 8; ++r) {
    float inv = 1.0f / lsum[r];
    size_t row = (size_t)b * TSEQ + qbase + m0 + r;
#pragma unroll
    for (int j = 0; j < 8; ++j) {
      attnh[row * D_MODEL + h * HEAD_DIM + j * 16 + lrow] =
          (_Float16)(o[j][r] * inv);
    }
  }
}

// ------------------- launcher -------------------
extern "C" void kernel_launch(void* const* d_in, const int* in_sizes, int n_in,
                              void* d_out, int out_size, void* d_ws, size_t ws_size,
                              hipStream_t stream) {
  const float* x  = (const float*)d_in[0];
  const float* Wq = (const float*)d_in[1];
  const float* Wk = (const float*)d_in[2];
  const float* Wv = (const float*)d_in[3];
  const float* Wo = (const float*)d_in[4];

  char* ws = (char*)d_ws;
  size_t off = 0;
  _Float16* xh    = (_Float16*)(ws + off); off += (size_t)ROWS * D_MODEL * 2;
  _Float16* WqT   = (_Float16*)(ws + off); off += (size_t)D_MODEL * D_MODEL * 2;
  _Float16* WkT   = (_Float16*)(ws + off); off += (size_t)HEAD_DIM * D_MODEL * 2;
  _Float16* WvT   = (_Float16*)(ws + off); off += (size_t)HEAD_DIM * D_MODEL * 2;
  _Float16* WoT   = (_Float16*)(ws + off); off += (size_t)D_MODEL * D_MODEL * 2;
  _Float16* qhb   = (_Float16*)(ws + off); off += (size_t)ROWS * D_MODEL * 2;
  _Float16* khb   = (_Float16*)(ws + off); off += (size_t)ROWS * HEAD_DIM * 2;
  _Float16* vTb   = (_Float16*)(ws + off); off += (size_t)ROWS * HEAD_DIM * 2;
  _Float16* attnh = (_Float16*)(ws + off); off += (size_t)ROWS * D_MODEL * 2;

  // 1) precision conversion + weight pre-transpose (f16, B-fragment friendly)
  {
    int n = ROWS * D_MODEL;
    convert_f32_to_f16<<<(n + 255) / 256, 256, 0, stream>>>(x, xh, n);
    int nw = D_MODEL * D_MODEL;
    convert_transpose_f16<<<(nw + 255) / 256, 256, 0, stream>>>(Wq, WqT, D_MODEL, D_MODEL);
    convert_transpose_f16<<<(nw + 255) / 256, 256, 0, stream>>>(Wo, WoT, D_MODEL, D_MODEL);
    int nk = D_MODEL * HEAD_DIM;
    convert_transpose_f16<<<(nk + 255) / 256, 256, 0, stream>>>(Wk, WkT, D_MODEL, HEAD_DIM);
    convert_transpose_f16<<<(nk + 255) / 256, 256, 0, stream>>>(Wv, WvT, D_MODEL, HEAD_DIM);
  }

  // 2) projections (64x32-per-wave register-blocked WMMA GEMMs)
  {
    int wavesQ = (ROWS / 64) * (D_MODEL / 32);        // 64*64 = 4096 waves
    gemm_wmma_blocked<<<wavesQ / 4, 128, 0, stream>>>(xh, WqT, qhb,
        ROWS, D_MODEL, D_MODEL, /*mode=*/0);
    int wavesKV = (ROWS / 64) * (HEAD_DIM / 32);      // 64*4 = 256 waves
    gemm_wmma_blocked<<<wavesKV / 4, 128, 0, stream>>>(xh, WkT, khb,
        ROWS, HEAD_DIM, D_MODEL, /*mode=*/1);
    gemm_wmma_blocked<<<wavesKV / 4, 128, 0, stream>>>(xh, WvT, vTb,
        ROWS, HEAD_DIM, D_MODEL, /*mode=*/2);
  }

  // 3) flash MQA attention (causal, shared KV head)
  {
    int wavesA = BSZ * N_HEADS * (TSEQ / 16);         // 4096 waves
    mqa_attn_kernel<<<wavesA / 4, 128, 0, stream>>>(qhb, khb, vTb, attnh);
  }

  // 4) output projection -> f32 d_out
  {
    int wavesO = (ROWS / 64) * (D_MODEL / 32);
    gemm_wmma_blocked<<<wavesO / 4, 128, 0, stream>>>(attnh, WoT, d_out,
        ROWS, D_MODEL, D_MODEL, /*mode=*/3);
  }
  (void)in_sizes; (void)n_in; (void)out_size; (void)ws_size;
}
